// DenseRNN_59957743452420
// MI455X (gfx1250) — compile-verified, matrix operand order checked by reading, not run on hardware
//
#include <hip/hip_runtime.h>
#include <stdint.h>

// ---------------- problem constants ----------------
#define BB   64
#define TT   2048
#define HH   512
#define OO   256
#define VV   32000
#define G3   1536           // 3*H

typedef __attribute__((ext_vector_type(8)))  float  v8f;
typedef __attribute__((ext_vector_type(16))) __bf16 v16bf;

union FragU { v16bf v; uint4 q[2]; };

// fp32 -> bf16 round-to-nearest-even
__device__ __forceinline__ uint16_t f32_to_bf16(float x) {
  uint32_t u = __float_as_uint(x);
  uint32_t r = u + 0x7FFFu + ((u >> 16) & 1u);
  return (uint16_t)(r >> 16);
}
__device__ __forceinline__ float bf16_to_f32(uint16_t h) {
  return __uint_as_float(((uint32_t)h) << 16);
}

// Load a 16x32 bf16 WMMA operand fragment (A-style; B uses the same pattern
// reading rows of the N-major weight matrix, since B = W^T column n == W row n).
// lane<16 : row = row0+lane,     K = k0 + {0..7, 16..23}
// lane>=16: row = row0+lane-16,  K = k0 + {8..15, 24..31}
__device__ __forceinline__ v16bf load_frag(const uint16_t* base, int ld, int row0, int k0) {
  const int lane = threadIdx.x & 31;
  const int r = row0 + (lane & 15);
  const int k = k0 + ((lane >> 4) << 3);
  const uint4* p = (const uint4*)(base + (size_t)r * (size_t)ld + (size_t)k);
  FragU f;
  f.q[0] = p[0];
  f.q[1] = p[2];   // +16 elements = +32 bytes
  return f.v;
}

__device__ __forceinline__ v8f wmma_bf16(v16bf a, v16bf b, v8f c) {
  return __builtin_amdgcn_wmma_f32_16x16x32_bf16(false, a, false, b, (short)0, c, false, false);
}

__device__ __forceinline__ float sigmoidf_(float x) {
  return 1.0f / (1.0f + __expf(-x));
}

// ---------------- kernel 1: f32 -> bf16 cast ----------------
__global__ void cast_f32_bf16_kernel(const float* __restrict__ src,
                                     uint16_t* __restrict__ dst, int n) {
  int i = blockIdx.x * blockDim.x + threadIdx.x;
  int stride = gridDim.x * blockDim.x;
  for (; i < n; i += stride) dst[i] = f32_to_bf16(src[i]);
}

// ---------------- kernel 2: P = emb @ w_ih^T  (bf16 out, no bias) ----------------
// grid (2000, 6), block 256. Wave w computes M-tile blockIdx.x, N cols
// [blockIdx.y*256 + w*32, +32) as two 16x16 tiles.
__global__ __launch_bounds__(256) void proj_emb_kernel(
    const uint16_t* __restrict__ embb,   // [V, H] bf16
    const uint16_t* __restrict__ wihb,   // [3H, H] bf16 (N-major)
    uint16_t* __restrict__ P)            // [V, 3H] bf16
{
  const int wave = threadIdx.x >> 5;
  const int lane = threadIdx.x & 31;
  const int mb = blockIdx.x * 16;
  const int n0 = blockIdx.y * 256 + wave * 32;

  v8f a0 = {}, a1 = {};
#pragma unroll 4
  for (int kb = 0; kb < 16; ++kb) {
    const int k0 = kb * 32;
    v16bf a  = load_frag(embb, HH, mb, k0);
    v16bf b0 = load_frag(wihb, HH, n0, k0);
    v16bf b1 = load_frag(wihb, HH, n0 + 16, k0);
    a0 = wmma_bf16(a, b0, a0);
    a1 = wmma_bf16(a, b1, a1);
  }
  const int rofs = (lane < 16) ? 0 : 8;
  const int c = lane & 15;
#pragma unroll
  for (int r = 0; r < 8; ++r) {
    const size_t row = (size_t)(mb + r + rofs) * G3;
    P[row + n0 + c]      = f32_to_bf16(a0[r]);
    P[row + n0 + 16 + c] = f32_to_bf16(a1[r]);
  }
}

// ---------------- global barrier (16 persistent WGs) ----------------
__device__ __forceinline__ void gbar(unsigned int* ctr, unsigned int target) {
  __threadfence();
  __syncthreads();
  if (threadIdx.x == 0) {
    __hip_atomic_fetch_add(ctr, 1u, __ATOMIC_RELEASE, __HIP_MEMORY_SCOPE_AGENT);
    __builtin_amdgcn_s_cluster_barrier();   // NOP if not cluster-dispatched
    while (__hip_atomic_load(ctr, __ATOMIC_RELAXED, __HIP_MEMORY_SCOPE_AGENT) < target) {
      __builtin_amdgcn_s_sleep(1);
    }
  }
  __syncthreads();
  __threadfence();
}

// ---------------- kernel 3: persistent GRU scan ----------------
// 16 workgroups x 256 threads. WG wg owns hidden cols [wg*32, wg*32+32).
// Wave w: M-tile mt = w/2 (batch rows 16*mt..), col-half jh = w&1 (16 cols).
// Its three accumulators are the r/z/n gate tiles for those exact 16 cols,
// so the GRU update is register-local. h ping-pongs through bf16 buffers in
// global memory (L2-resident, 64KB each); fp32 master h stays in registers.
// gi gathers are issued BEFORE the WMMA K-loop so their L2 latency hides
// under the GEMM (loads complete in issue order).
__global__ __launch_bounds__(256) void gru_scan_kernel(
    const int*      __restrict__ x,      // [B, T]
    const float*    __restrict__ h0,     // [B, H]
    const uint16_t* __restrict__ whh,    // [3H, H] bf16
    const uint16_t* __restrict__ P,      // [V, 3H] bf16
    const float*    __restrict__ b_ih,   // [3H]
    const float*    __restrict__ b_hh,   // [3H]
    uint16_t*       __restrict__ hbuf,   // 2 x [B*H] bf16 (ping-pong)
    float*          __restrict__ out_h,  // [B, H] fp32 (d_out + B*O)
    unsigned int*   __restrict__ ctr)    // barrier counter (zeroed by memset)
{
  __shared__ int xs[BB];

  const int wg   = blockIdx.x;           // 0..15
  const int tid  = threadIdx.x;
  const int wave = tid >> 5;
  const int lane = tid & 31;
  const int mt   = wave >> 1;            // 0..3
  const int jh   = wave & 1;             // 0..1
  const int mb   = mt * 16;
  const int j0   = wg * 32 + jh * 16;    // base hidden column (16 cols)
  const int col  = j0 + (lane & 15);
  const int rofs = (lane < 16) ? 0 : 8;

  // per-lane fused biases
  const float cb_r = b_ih[col]        + b_hh[col];
  const float cb_z = b_ih[HH  + col]  + b_hh[HH  + col];
  const float bi_n = b_ih[2*HH + col];
  const float bh_n = b_hh[2*HH + col];

  uint16_t* hb0 = hbuf;
  uint16_t* hb1 = hbuf + BB * HH;

  // fp32 master copy of this wave's h block; seed from h0, publish bf16.
  v8f hm;
#pragma unroll
  for (int r = 0; r < 8; ++r) {
    const int m = mb + r + rofs;
    hm[r] = h0[m * HH + col];
    hb0[m * HH + col] = f32_to_bf16(hm[r]);
  }

  unsigned int bar = 0;
  gbar(ctr, 16u * (++bar));              // h0 published everywhere

  for (int t = 0; t < TT; ++t) {
    const uint16_t* hin  = (t & 1) ? hb1 : hb0;
    uint16_t*       hout = (t & 1) ? hb0 : hb1;

    if (tid < BB) xs[tid] = x[tid * TT + t];
    __syncthreads();

    // ---- issue gi gathers first: L2 latency hides under the GEMM below ----
    float gir[8], giz[8], gin[8];
#pragma unroll
    for (int r = 0; r < 8; ++r) {
      const uint16_t* pr = P + (size_t)xs[mb + r + rofs] * G3 + col;
      gir[r] = bf16_to_f32(pr[0]);
      giz[r] = bf16_to_f32(pr[HH]);
      gin[r] = bf16_to_f32(pr[2 * HH]);
    }

    // gh tiles: [16 batch x 16 cols] for each gate, K = 512
    v8f ar = {}, az = {}, an = {};
#pragma unroll 4
    for (int kb = 0; kb < 16; ++kb) {
      const int k0 = kb * 32;
      v16bf a  = load_frag(hin, HH, mb, k0);
      v16bf br = load_frag(whh, HH, j0,          k0);
      v16bf bz = load_frag(whh, HH, HH   + j0,   k0);
      v16bf bn = load_frag(whh, HH, 2*HH + j0,   k0);
      ar = wmma_bf16(a, br, ar);
      az = wmma_bf16(a, bz, az);
      an = wmma_bf16(a, bn, an);
    }

    // elementwise GRU update (all register-local)
#pragma unroll
    for (int r = 0; r < 8; ++r) {
      const int m  = mb + r + rofs;
      const float rg = sigmoidf_(gir[r] + cb_r + ar[r]);
      const float zg = sigmoidf_(giz[r] + cb_z + az[r]);
      const float ng = tanhf(gin[r] + bi_n + rg * (an[r] + bh_n));
      const float hn = (1.0f - zg) * ng + zg * hm[r];
      hm[r] = hn;
      hout[m * HH + col] = f32_to_bf16(hn);
    }

    gbar(ctr, 16u * (++bar));            // publish h_{t+1} to all WGPs
  }

  // final fp32 hidden state straight from registers
#pragma unroll
  for (int r = 0; r < 8; ++r) {
    const int m = mb + r + rofs;
    out_h[m * HH + col] = hm[r];
  }
}

// ---------------- kernel 4: y = h_last @ w_out^T + b_out ----------------
// grid (4, 2), block 256. Wave w: M-tile blockIdx.x, N-tile blockIdx.y*8 + w.
__global__ __launch_bounds__(256) void out_gemm_kernel(
    const uint16_t* __restrict__ hlast,   // [B, H] bf16 (ping-pong buf 0: T even)
    const uint16_t* __restrict__ woutb,   // [O, H] bf16
    const float*    __restrict__ b_out,   // [O]
    float*          __restrict__ y)       // [B, O]
{
  const int wave = threadIdx.x >> 5;
  const int lane = threadIdx.x & 31;
  const int mb = blockIdx.x * 16;
  const int n0 = (blockIdx.y * 8 + wave) * 16;

  v8f acc = {};
#pragma unroll 4
  for (int kb = 0; kb < 16; ++kb) {
    const int k0 = kb * 32;
    v16bf a = load_frag(hlast, HH, mb, k0);
    v16bf b = load_frag(woutb, HH, n0, k0);
    acc = wmma_bf16(a, b, acc);
  }
  const int rofs = (lane < 16) ? 0 : 8;
  const int c = n0 + (lane & 15);
  const float bias = b_out[c];
#pragma unroll
  for (int r = 0; r < 8; ++r) {
    y[(mb + r + rofs) * OO + c] = acc[r] + bias;
  }
}

// ---------------- host glue ----------------
static inline size_t align256(size_t v) { return (v + 255) & ~(size_t)255; }

extern "C" void kernel_launch(void* const* d_in, const int* in_sizes, int n_in,
                              void* d_out, int out_size, void* d_ws, size_t ws_size,
                              hipStream_t stream) {
  (void)in_sizes; (void)n_in; (void)out_size; (void)ws_size;

  const int*   x     = (const int*)  d_in[0];
  const float* h0    = (const float*)d_in[1];
  const float* emb   = (const float*)d_in[2];
  const float* w_ih  = (const float*)d_in[3];
  const float* w_hh  = (const float*)d_in[4];
  const float* b_ih  = (const float*)d_in[5];
  const float* b_hh  = (const float*)d_in[6];
  const float* w_out = (const float*)d_in[7];
  const float* b_out = (const float*)d_in[8];

  // workspace carve-up (bf16 copies + projected table + h ping-pong + barrier)
  char* ws = (char*)d_ws;
  size_t off = 0;
  uint16_t* emb_bf  = (uint16_t*)(ws + off); off = align256(off + (size_t)VV * HH * 2);
  uint16_t* wih_bf  = (uint16_t*)(ws + off); off = align256(off + (size_t)G3 * HH * 2);
  uint16_t* whh_bf  = (uint16_t*)(ws + off); off = align256(off + (size_t)G3 * HH * 2);
  uint16_t* wout_bf = (uint16_t*)(ws + off); off = align256(off + (size_t)OO * HH * 2);
  uint16_t* P       = (uint16_t*)(ws + off); off = align256(off + (size_t)VV * G3 * 2);
  uint16_t* hbuf    = (uint16_t*)(ws + off); off = align256(off + (size_t)2 * BB * HH * 2);
  unsigned int* ctr = (unsigned int*)(ws + off); off = align256(off + 256);

  float* y_out = (float*)d_out;                 // [B, O]
  float* h_out = (float*)d_out + BB * OO;       // [B, H]

  // 1) casts to bf16
  cast_f32_bf16_kernel<<<8192, 256, 0, stream>>>(emb,   emb_bf,  VV * HH);
  cast_f32_bf16_kernel<<<1024, 256, 0, stream>>>(w_ih,  wih_bf,  G3 * HH);
  cast_f32_bf16_kernel<<<1024, 256, 0, stream>>>(w_hh,  whh_bf,  G3 * HH);
  cast_f32_bf16_kernel<<<256,  256, 0, stream>>>(w_out, wout_bf, OO * HH);

  // 2) projected embedding table P = emb @ w_ih^T   (V x 3H, bf16, L2-resident)
  proj_emb_kernel<<<dim3(VV / 16, G3 / 256), 256, 0, stream>>>(emb_bf, wih_bf, P);

  // 3) barrier counter -> 0, then persistent scan over T (16 WGPs)
  hipMemsetAsync(ctr, 0, 256, stream);
  gru_scan_kernel<<<16, 256, 0, stream>>>(x, h0, whh_bf, P, b_ih, b_hh,
                                          hbuf, h_out, ctr);

  // 4) output projection; final h lives in ping-pong buffer 0 (T even)
  out_gemm_kernel<<<dim3(BB / 16, OO / 128), 256, 0, stream>>>(hbuf, wout_bf,
                                                               b_out, y_out);
}
